// LinkPredictor_22703197127228
// MI455X (gfx1250) — matrix operand order repaired and checked by reading, not compiled, not run on hardware
//
#include <hip/hip_runtime.h>
#include <hip/hip_bf16.h>
#include <stdint.h>

typedef __attribute__((ext_vector_type(16))) __bf16    v16bf;
typedef __attribute__((ext_vector_type(8)))  float     v8f;
typedef __attribute__((ext_vector_type(4)))  uint32_t  u32x4;

union ABfrag {
  v16bf v;
  struct { u32x4 q0, q1; };
};

#define D     256
#define MBLK  64
#define SX    520   // X / H1 LDS row stride in bf16 elems (512 + 8 pad -> 1040B, 16B aligned)
#define SH    264   // H2 / H3 LDS row stride (256 + 8 pad -> 528B, 16B aligned)

// ---------------- f32 -> bf16 conversion pre-pass ----------------
__global__ void cvt_f32_to_bf16(const float* __restrict__ src,
                                __bf16* __restrict__ dst, int n) {
  int i = blockIdx.x * blockDim.x + threadIdx.x;
  if (i < n) dst[i] = (__bf16)src[i];
}

// A fragment: 16x32 bf16 from LDS. lanes 0-15: K = kc*32 + {0..7 | 16..23},
//                                  lanes16-31: K = kc*32 + {8..15 | 24..31}
__device__ __forceinline__ void loadA(ABfrag& a, const __bf16* Xs, int sX,
                                      int mBase, int kc, int l15, int half) {
  const __bf16* ap = Xs + (size_t)(mBase + l15) * sX + kc * 32 + half * 8;
  a.q0 = *(const u32x4*)(ap);
  a.q1 = *(const u32x4*)(ap + 16);
}

// B fragment: 32x16 bf16, column n = l15; lanes 0-15 hold K {0..7 | 16..23}, 16-31 hold K {8..15 | 24..31}.
// W is row-major [N][K] -> two 16-byte chunks per lane (K 0..7 and 16..23 of its half), no transpose needed.
__device__ __forceinline__ void loadB(ABfrag& b, const __bf16* W, int K,
                                      int nBase, int j, int kc, int l15, int half) {
  const __bf16* bp = W + (size_t)(nBase + j * 16 + l15) * K + kc * 32 + half * 8;
  b.q0 = *(const u32x4*)(bp);
  b.q1 = *(const u32x4*)(bp + 16);
}

// ---------------- tiled WMMA GEMM helper ----------------
// One wave computes a [32 x 64] output tile (2 M-tiles x 4 N-tiles):
//   Ho[mBase+0..31, nBase+0..63] = relu( Xs[...] @ W[...]^T + bias )
// Software-pipelined: B prefetched one tile ahead, A one kc ahead.
__device__ __forceinline__ void mlp_gemm_2x4(const __bf16* __restrict__ Xs, int sX,
                                             const __bf16* __restrict__ W,
                                             const float* __restrict__ bias, int K,
                                             __bf16* __restrict__ Ho, int sO,
                                             int mBase, int nBase, int lane) {
  const int l15  = lane & 15;
  const int half = lane >> 4;
  const int kch  = K >> 5;

  v8f acc[2][4];
#pragma unroll
  for (int j = 0; j < 4; ++j) {
    float bv = bias[nBase + j * 16 + l15];
    acc[0][j] = (v8f){bv, bv, bv, bv, bv, bv, bv, bv};
    acc[1][j] = acc[0][j];
  }

  ABfrag a[2], an[2], bc, bn;
  loadA(a[0], Xs, sX, mBase,      0, l15, half);
  loadA(a[1], Xs, sX, mBase + 16, 0, l15, half);
  loadB(bc, W, K, nBase, 0, 0, l15, half);

#pragma unroll
  for (int kc = 0; kc < kch; ++kc) {
#pragma unroll
    for (int j = 0; j < 4; ++j) {
      // prefetch next B tile (one ahead)
      const int jn = (j + 1) & 3;
      const int kn = (j == 3) ? kc + 1 : kc;
      if (kn < kch) loadB(bn, W, K, nBase, jn, kn, l15, half);
      // prefetch next kc's A fragments early in the j-chain
      if (j == 0 && kc + 1 < kch) {
        loadA(an[0], Xs, sX, mBase,      kc + 1, l15, half);
        loadA(an[1], Xs, sX, mBase + 16, kc + 1, l15, half);
      }
      acc[0][j] = __builtin_amdgcn_wmma_f32_16x16x32_bf16(
          false, a[0].v, false, bc.v, (short)0, acc[0][j], false, false);
      acc[1][j] = __builtin_amdgcn_wmma_f32_16x16x32_bf16(
          false, a[1].v, false, bc.v, (short)0, acc[1][j], false, false);
      bc = bn;
    }
    a[0] = an[0];
    a[1] = an[1];
  }

  // ReLU + bf16 convert + scatter to LDS.
  // C layout: VGPR r, lane l -> row mBase + r + 8*(l>>4), col nBase + (l&15)
#pragma unroll
  for (int t = 0; t < 2; ++t) {
#pragma unroll
    for (int j = 0; j < 4; ++j) {
#pragma unroll
      for (int r = 0; r < 8; ++r) {
        float v = acc[t][j][r];
        v = v > 0.0f ? v : 0.0f;
        Ho[(size_t)(mBase + t * 16 + r + half * 8) * sO + nBase + j * 16 + l15] = (__bf16)v;
      }
    }
  }
}

// ---------------- main fused MLP kernel ----------------
__global__ __launch_bounds__(256, 4) void linkpred_mlp(
    const __bf16* __restrict__ zb, const long long* __restrict__ eidx,
    const __bf16* __restrict__ W1b, const float* __restrict__ b1,
    const __bf16* __restrict__ W2b, const float* __restrict__ b2,
    const __bf16* __restrict__ W3b, const float* __restrict__ b3,
    const float*  __restrict__ W4f, const float* __restrict__ b4,
    float* __restrict__ out, int nEdges) {
  __shared__ __bf16 smA[MBLK * SX];   // X, later aliased as H2
  __shared__ __bf16 smB[MBLK * SX];   // H1, later aliased as H3

  __bf16* Xs  = smA;
  __bf16* H1s = smB;
  __bf16* H2s = smA;                  // layer2 output overwrites X
  __bf16* H3s = smB;                  // layer3 output overwrites H1

  const int tid  = threadIdx.x;
  const int wave = tid >> 5;
  const int lane = tid & 31;
  const long long eBase = (long long)blockIdx.x * MBLK;

  // ---- stage gathered [z[row] || z[col]] -> X [64 x 512] bf16 ----
  {
    int i = tid >> 2;                 // edge within block
    int q = tid & 3;                  // quarter of the 512-elem row
    long long e = eBase + i;
    if (e >= nEdges) e = 0;
    long long node = (q < 2) ? eidx[e] : eidx[(long long)nEdges + e];
    const __bf16* src = zb + node * D + (q & 1) * 128;
    __bf16*       dst = Xs + (size_t)i * SX + q * 128;
#pragma unroll
    for (int c = 0; c < 16; ++c)
      *(u32x4*)(dst + c * 8) = *(const u32x4*)(src + c * 8);
  }
  __syncthreads();

  const int mB = (wave & 1) * 32;     // two 16-row M-tiles per wave
  const int ng = wave >> 1;           // 0..3 : N partition

  // ---- layer 1: X[64x512] @ W1^T -> relu -> H1[64x512] ----
#pragma unroll
  for (int g = 0; g < 2; ++g)
    mlp_gemm_2x4(Xs, SX, W1b, b1, 512, H1s, SX, mB, ng * 128 + g * 64, lane);
  __syncthreads();

  // ---- layer 2: H1[64x512] @ W2^T -> relu -> H2[64x256] ----
  mlp_gemm_2x4(H1s, SX, W2b, b2, 512, H2s, SH, mB, ng * 64, lane);
  __syncthreads();

  // ---- layer 3: H2[64x256] @ W3^T -> relu -> H3[64x256] ----
  mlp_gemm_2x4(H2s, SH, W3b, b3, 256, H3s, SH, mB, ng * 64, lane);
  __syncthreads();

  // ---- layer 4: H3[64x256] . W4 + b4 -> sigmoid ----
  {
    int m = tid >> 2;                 // edge within block
    int p = tid & 3;                  // quarter of the 256-dot
    const __bf16* hp = H3s + (size_t)m * SH + p * 64;
    const float*  wp = W4f + p * 64;
    float sum = 0.0f;
#pragma unroll 8
    for (int k = 0; k < 64; ++k) sum += (float)hp[k] * wp[k];
    sum += __shfl_xor(sum, 1, 32);
    sum += __shfl_xor(sum, 2, 32);
    if (p == 0) {
      long long e = eBase + m;
      if (e < nEdges) {
        float x = sum + b4[0];
        out[e] = 1.0f / (1.0f + __expf(-x));
      }
    }
  }
}

// ---------------- launch ----------------
extern "C" void kernel_launch(void* const* d_in, const int* in_sizes, int n_in,
                              void* d_out, int out_size, void* d_ws, size_t ws_size,
                              hipStream_t stream) {
  const float*     z    = (const float*)d_in[0];
  const long long* eidx = (const long long*)d_in[1];
  const float*     W1   = (const float*)d_in[2];
  const float*     b1   = (const float*)d_in[3];
  const float*     W2   = (const float*)d_in[4];
  const float*     b2   = (const float*)d_in[5];
  const float*     W3   = (const float*)d_in[6];
  const float*     b3   = (const float*)d_in[7];
  const float*     W4   = (const float*)d_in[8];
  const float*     b4   = (const float*)d_in[9];
  float*           out  = (float*)d_out;

  const int nZ     = in_sizes[0];       // nNodes * 256
  const int nEdges = in_sizes[1] / 2;

  char* ws = (char*)d_ws;
  __bf16* zb  = (__bf16*)ws;
  __bf16* W1b = (__bf16*)(ws + (size_t)nZ * 2);
  __bf16* W2b = (__bf16*)((char*)W1b + 512 * 512 * 2);
  __bf16* W3b = (__bf16*)((char*)W2b + 256 * 512 * 2);

  cvt_f32_to_bf16<<<(nZ + 255) / 256, 256, 0, stream>>>(z, zb, nZ);
  cvt_f32_to_bf16<<<(512 * 512 + 255) / 256, 256, 0, stream>>>(W1, W1b, 512 * 512);
  cvt_f32_to_bf16<<<(256 * 512 + 255) / 256, 256, 0, stream>>>(W2, W2b, 256 * 512);
  cvt_f32_to_bf16<<<(256 * 256 + 255) / 256, 256, 0, stream>>>(W3, W3b, 256 * 256);

  const int grid = (nEdges + MBLK - 1) / MBLK;
  linkpred_mlp<<<grid, 256, 0, stream>>>(zb, eidx, W1b, b1, W2b, b2,
                                         W3b, b3, W4, b4, out, nEdges);
}
